// Model_2877628088432
// MI455X (gfx1250) — compile-verified, matrix operand order checked by reading, not run on hardware
//
#include <hip/hip_runtime.h>

// ---------------------------------------------------------------------------
// Reformer (LSH attention) forward for MI455X / gfx1250, wave32 + WMMA bf16.
// ---------------------------------------------------------------------------

typedef __bf16 bf16_t;
typedef __attribute__((ext_vector_type(16))) __bf16 v16bf;
typedef __attribute__((ext_vector_type(8)))  float  v8f;

#define B_      32
#define SEQ_    720
#define PRED_   720
#define ENCIN_  7
#define COUT_   7
#define LABEL_  48
#define D_      512
#define H_      8
#define DH_     64
#define DFF_    2048
#define EL_     2
#define NH_     4
#define NB_     360
#define N_      1440            // SEQ + PRED
#define BH_     (B_ * H_)       // 256
#define NJ_     (NH_ * N_)      // 5760 keys per bh row
#define CCH_    (NJ_ / 4)       // 1440 chunks of BUCKET=4
#define ROWS_   (B_ * N_)       // 46080 token rows

__device__ __forceinline__ bf16_t f2bf(float f) {
  unsigned u = __builtin_bit_cast(unsigned, f);
  unsigned r = (u + 0x7FFFu + ((u >> 16) & 1u)) >> 16;
  unsigned short s = (unsigned short)r;
  return __builtin_bit_cast(bf16_t, s);
}

// --------------------------- elementwise convert ---------------------------
__global__ void k_convert_bf16(const float* __restrict__ in, bf16_t* __restrict__ out,
                               long long n) {
  long long i = (long long)blockIdx.x * blockDim.x + threadIdx.x;
  if (i < n) out[i] = f2bf(in[i]);
}

// ------------------------------- embedding ---------------------------------
__global__ void k_embed(const float* __restrict__ xe,  const float* __restrict__ xme,
                        const float* __restrict__ xd,  const float* __restrict__ xmd,
                        const float* __restrict__ convw, const float* __restrict__ timew,
                        float* __restrict__ h, bf16_t* __restrict__ hb) {
  long long idx = (long long)blockIdx.x * blockDim.x + threadIdx.x;
  if (idx >= (long long)ROWS_ * D_) return;
  int d = (int)(idx % D_);
  int t = (int)((idx / D_) % N_);
  int b = (int)(idx / ((long long)N_ * D_));

  float tok = 0.f;
#pragma unroll
  for (int k = 0; k < 3; ++k) {
    int s = t + k - 1;
    s = (s + N_) % N_;
#pragma unroll
    for (int c = 0; c < ENCIN_; ++c) {
      float xv = (s < SEQ_)
        ? xe[((size_t)b * SEQ_ + s) * ENCIN_ + c]
        : xd[((size_t)b * (LABEL_ + PRED_) + LABEL_ + (s - SEQ_)) * ENCIN_ + c];
      tok += xv * convw[((size_t)k * ENCIN_ + c) * D_ + d];
    }
  }
  int i2 = d >> 1;
  float div = __expf((float)(2 * i2) * (-9.210340371976184f / (float)D_));
  float ang = (float)t * div;
  float pe = (d & 1) ? __cosf(ang) : __sinf(ang);
  float tf = 0.f;
#pragma unroll
  for (int q = 0; q < 4; ++q) {
    float mv = (t < SEQ_)
      ? xme[((size_t)b * SEQ_ + t) * 4 + q]
      : xmd[((size_t)b * (LABEL_ + PRED_) + LABEL_ + (t - SEQ_)) * 4 + q];
    tf += mv * timew[(size_t)q * D_ + d];
  }
  float v = tok + pe + tf;
  h[idx] = v;
  hb[idx] = f2bf(v);
}

// --------------------------- WMMA bf16 GEMM --------------------------------
// C[M,N] = A[M,K] @ B[K,N] (+bias)(gelu)(bf16 mirror). Row-major. M%128==0,
// N%64==0, K%32==0. Block = 256 threads = 8 wave32s; block tile 128x64; each
// wave owns a 32x32 region = 4 WMMA tiles per K-step (2 A-frags x 2 B-frags).
// B tile is stored TRANSPOSED in LDS so both A and B fragments are two
// contiguous ds_load_b128 per lane (ISA 7.12.2 16-bit layout).
#define GF_BIAS 1
#define GF_GELU 2
#define GF_BF16 4
#define BM_ 128
#define BN_ 64
#define BK_ 32
#define LDP_ (BK_ + 8)          // padded row: 40 elems = 80B (16B aligned)
__global__ __launch_bounds__(256)
void k_gemm_bf16(const bf16_t* __restrict__ A, const bf16_t* __restrict__ Bw,
                 float* __restrict__ C, bf16_t* __restrict__ Cb,
                 const float* __restrict__ bias,
                 int M, int Ncols, int K, int flags) {
  __shared__ bf16_t lA[BM_][LDP_];    // 128*40*2 = 10 KB
  __shared__ bf16_t lBt[BN_][LDP_];   //  64*40*2 =  5 KB
  const int tid   = threadIdx.x;
  const int lane  = tid & 31;
  const int wave  = tid >> 5;
  const int waveM = wave >> 1;               // 0..3 -> M offset *32
  const int waveN = wave & 1;                // 0..1 -> N offset *32
  const int blockM = blockIdx.y * BM_;
  const int blockN = blockIdx.x * BN_;

  // A cooperative load: 128x32 elems; thread -> row ar, 16 contiguous elems.
  const int ar = tid >> 1;                   // 0..127
  const int ac = (tid & 1) * 16;             // 0 or 16
  // B cooperative load+transpose: 2(K) x 4(N) sub-block per thread.
  const int kp = tid >> 4;                   // 0..15 -> k rows 2*kp, 2*kp+1
  const int cg = tid & 15;                   // 0..15 -> cols cg*4..+3

  const int fr   = lane & 15;
  const int half = lane >> 4;

  v8f acc[2][2] = {{{}, {}}, {{}, {}}};

  for (int k0 = 0; k0 < K; k0 += BK_) {
    // ---- stage A tile ----
    const bf16_t* Ap = A + (size_t)(blockM + ar) * K + k0 + ac;
    *(uint4*)&lA[ar][ac]     = *(const uint4*)Ap;
    *(uint4*)&lA[ar][ac + 8] = *(const uint4*)(Ap + 8);
    // ---- stage B tile transposed, packing (k,k+1) pairs as u32 ----
    const int kk = kp * 2;
    const bf16_t* Bp = Bw + (size_t)(k0 + kk) * Ncols + blockN + cg * 4;
    uint2 r0 = *(const uint2*)Bp;
    uint2 r1 = *(const uint2*)(Bp + Ncols);
    const unsigned short* p0 = (const unsigned short*)&r0;
    const unsigned short* p1 = (const unsigned short*)&r1;
#pragma unroll
    for (int i = 0; i < 4; ++i) {
      unsigned pack = (unsigned)p0[i] | ((unsigned)p1[i] << 16);
      *(unsigned*)&lBt[cg * 4 + i][kk] = pack;
    }
    if (k0 + BK_ < K) {      // pull next K-tile toward cache (global_prefetch_b8)
      __builtin_prefetch(A + (size_t)(blockM + ar) * K + (k0 + BK_) + ac, 0, 0);
      __builtin_prefetch(Bw + (size_t)(k0 + BK_ + kk) * Ncols + blockN + cg * 4, 0, 0);
    }
    __syncthreads();

    // ---- fragments: two contiguous b128 reads each (ISA 16-bit layout) ----
    v16bf afrag[2], bfrag[2];
#pragma unroll
    for (int m = 0; m < 2; ++m) {
      const bf16_t* src = &lA[waveM * 32 + m * 16 + fr][half * 8];
      ((uint4*)&afrag[m])[0] = *(const uint4*)src;
      ((uint4*)&afrag[m])[1] = *(const uint4*)(src + 16);
    }
#pragma unroll
    for (int nn = 0; nn < 2; ++nn) {
      const bf16_t* src = &lBt[waveN * 32 + nn * 16 + fr][half * 8];
      ((uint4*)&bfrag[nn])[0] = *(const uint4*)src;
      ((uint4*)&bfrag[nn])[1] = *(const uint4*)(src + 16);
    }
#pragma unroll
    for (int m = 0; m < 2; ++m)
#pragma unroll
      for (int nn = 0; nn < 2; ++nn)
        acc[m][nn] = __builtin_amdgcn_wmma_f32_16x16x32_bf16(
            false, afrag[m], false, bfrag[nn], (short)0, acc[m][nn], false, false);
    __syncthreads();
  }

  // ---- epilogue ----
#pragma unroll
  for (int m = 0; m < 2; ++m) {
#pragma unroll
    for (int nn = 0; nn < 2; ++nn) {
      const int gn = blockN + waveN * 32 + nn * 16 + fr;
      float bv = (flags & GF_BIAS) ? bias[gn] : 0.f;
#pragma unroll
      for (int r = 0; r < 8; ++r) {
        const int gm = blockM + waveM * 32 + m * 16 + half * 8 + r;
        float v = acc[m][nn][r] + bv;
        if (flags & GF_GELU) v = 0.5f * v * (1.0f + erff(v * 0.70710678118f));
        C[(size_t)gm * Ncols + gn] = v;
        if (flags & GF_BF16) Cb[(size_t)gm * Ncols + gn] = f2bf(v);
      }
    }
  }
}

// ------------------------------ LSH buckets --------------------------------
__global__ void k_buckets(const float* __restrict__ qk, const float* __restrict__ rot,
                          int* __restrict__ buckets) {
  long long idx = (long long)blockIdx.x * blockDim.x + threadIdx.x;
  if (idx >= (long long)BH_ * NH_ * N_) return;
  int t  = (int)(idx % N_);
  int nh = (int)((idx / N_) % NH_);
  int bh = (int)(idx / ((long long)NH_ * N_));
  int b = bh >> 3, hh = bh & 7;
  const float* q = qk + ((size_t)b * N_ + t) * D_ + hh * DH_;
  float qr[DH_];
#pragma unroll
  for (int d = 0; d < DH_; ++d) qr[d] = q[d];

  float best = -3.4e38f; int bi = 0;
  for (int i = 0; i < NB_ / 2; ++i) {
    float s = 0.f;
#pragma unroll
    for (int d = 0; d < DH_; ++d)
      s += qr[d] * rot[((size_t)d * NH_ + nh) * (NB_ / 2) + i];
    if (s > best)  { best = s;  bi = i; }
    if (-s > best) { best = -s; bi = i + NB_ / 2; }
  }
  buckets[(size_t)bh * NJ_ + (size_t)nh * N_ + t] = bi + nh * NB_;
}

// ------------------------ stable counting sort per bh ----------------------
__global__ __launch_bounds__(256)
void k_sort(const int* __restrict__ buckets, int* __restrict__ order,
            int* __restrict__ undo) {
  __shared__ unsigned hist[NH_ * NB_];
  __shared__ unsigned startb[NH_ * NB_];
  const int bh = blockIdx.x, tid = threadIdx.x;
  for (int i = tid; i < NH_ * NB_; i += 256) hist[i] = 0;
  __syncthreads();
  for (int j = tid; j < NJ_; j += 256)
    atomicAdd(&hist[buckets[(size_t)bh * NJ_ + j]], 1u);
  __syncthreads();
  if (tid == 0) {
    unsigned run = 0;
    for (int i = 0; i < NH_ * NB_; ++i) { startb[i] = run; run += hist[i]; }
  }
  __syncthreads();
  for (int i = tid; i < NH_ * NB_; i += 256) hist[i] = 0;
  __syncthreads();
  if (tid < NH_) {   // hash rounds own disjoint bucket ranges: no collisions
    const int nh = tid;
    for (int t = 0; t < N_; ++t) {
      int j = nh * N_ + t;
      int bk = buckets[(size_t)bh * NJ_ + j];
      unsigned pos = startb[bk] + hist[bk]++;
      order[(size_t)bh * NJ_ + pos] = j;
      undo[(size_t)bh * NJ_ + j] = (int)pos;
    }
  }
}

// ------------------------- chunked LSH attention ---------------------------
__global__ void k_attn(const float* __restrict__ qk, const float* __restrict__ vbuf,
                       const int* __restrict__ order,
                       float* __restrict__ so, float* __restrict__ slog) {
  long long idx = (long long)blockIdx.x * blockDim.x + threadIdx.x;
  if (idx >= (long long)BH_ * NJ_) return;
  const int p  = (int)(idx % NJ_);
  const int bh = (int)(idx / NJ_);
  const int b = bh >> 3, hh = bh & 7;
  const int j = order[(size_t)bh * NJ_ + p];
  const int t = j % N_;

  const float* qrow = qk + ((size_t)b * N_ + t) * D_ + hh * DH_;
  float q[DH_];
#pragma unroll
  for (int d = 0; d < DH_; ++d) q[d] = qrow[d];

  const int c  = p >> 2;
  const int pc = (c + CCH_ - 1) % CCH_;
  float dots[8]; int tks[8];
#pragma unroll
  for (int kk = 0; kk < 8; ++kk) {
    const int pk = (kk < 4) ? (c * 4 + kk) : (pc * 4 + (kk - 4));
    const int jk = order[(size_t)bh * NJ_ + pk];
    const int tk = jk % N_;
    tks[kk] = tk;
    const float* krow = qk + ((size_t)b * N_ + tk) * D_ + hh * DH_;
    float s2 = 0.f, dt = 0.f;
#pragma unroll
    for (int d = 0; d < DH_; ++d) { float kv = krow[d]; s2 += kv * kv; }
    float inv = 1.0f / fmaxf(__fsqrt_rn(s2), 1e-12f);
#pragma unroll
    for (int d = 0; d < DH_; ++d) dt += q[d] * krow[d];
    dt *= inv * 0.125f;
    if (t == tk) dt = -5e4f;
    dots[kk] = dt;
  }
  float mx = dots[0];
#pragma unroll
  for (int kk = 1; kk < 8; ++kk) mx = fmaxf(mx, dots[kk]);
  float ssum = 0.f;
#pragma unroll
  for (int kk = 0; kk < 8; ++kk) ssum += __expf(dots[kk] - mx);
  const float lse = mx + __logf(ssum);

  float o[DH_];
#pragma unroll
  for (int d = 0; d < DH_; ++d) o[d] = 0.f;
#pragma unroll
  for (int kk = 0; kk < 8; ++kk) {
    const float w = __expf(dots[kk] - lse);
    const float* vr = vbuf + ((size_t)b * N_ + tks[kk]) * D_ + hh * DH_;
#pragma unroll
    for (int d = 0; d < DH_; ++d) o[d] += w * vr[d];
  }
  float* dst = so + ((size_t)bh * NJ_ + j) * DH_;
#pragma unroll
  for (int d = 0; d < DH_; ++d) dst[d] = o[d];
  slog[(size_t)bh * NJ_ + j] = lse;
}

// ------------------- combine hash rounds (softmax over lse) ----------------
__global__ void k_combine(const float* __restrict__ so, const float* __restrict__ slog,
                          float* __restrict__ attn) {
  long long idx = (long long)blockIdx.x * blockDim.x + threadIdx.x;
  if (idx >= (long long)BH_ * N_) return;
  const int t  = (int)(idx % N_);
  const int bh = (int)(idx / N_);
  const int b = bh >> 3, hh = bh & 7;
  float l[NH_];
#pragma unroll
  for (int nh = 0; nh < NH_; ++nh) l[nh] = slog[(size_t)bh * NJ_ + nh * N_ + t];
  float mx = l[0];
#pragma unroll
  for (int nh = 1; nh < NH_; ++nh) mx = fmaxf(mx, l[nh]);
  float s = 0.f;
#pragma unroll
  for (int nh = 0; nh < NH_; ++nh) { l[nh] = __expf(l[nh] - mx); s += l[nh]; }
  const float inv = 1.0f / s;
  float* dst = attn + ((size_t)b * N_ + t) * D_ + hh * DH_;
#pragma unroll
  for (int d = 0; d < DH_; ++d) {
    float acc = 0.f;
#pragma unroll
    for (int nh = 0; nh < NH_; ++nh)
      acc += l[nh] * so[((size_t)bh * NJ_ + nh * N_ + t) * DH_ + d];
    dst[d] = acc * inv;
  }
}

// -------------------- LayerNorm (+residual, fused bf16) --------------------
__global__ __launch_bounds__(256)
void k_ln(const float* __restrict__ x, const float* __restrict__ resid,
          const float* __restrict__ g, const float* __restrict__ bta,
          float* __restrict__ out, bf16_t* __restrict__ outb) {
  const int row = blockIdx.x, tid = threadIdx.x;
  __shared__ float s1[256], s2[256];
  const size_t base = (size_t)row * D_;
  float x0 = x[base + tid];
  float x1 = x[base + tid + 256];
  if (resid) { x0 += resid[base + tid]; x1 += resid[base + tid + 256]; }
  s1[tid] = x0 + x1;
  s2[tid] = x0 * x0 + x1 * x1;
  __syncthreads();
  for (int s = 128; s > 0; s >>= 1) {
    if (tid < s) { s1[tid] += s1[tid + s]; s2[tid] += s2[tid + s]; }
    __syncthreads();
  }
  const float mean = s1[0] * (1.0f / D_);
  const float var  = s2[0] * (1.0f / D_) - mean * mean;
  const float inv  = rsqrtf(var + 1e-5f);
  float y0 = (x0 - mean) * inv * g[tid]       + bta[tid];
  float y1 = (x1 - mean) * inv * g[tid + 256] + bta[tid + 256];
  out[base + tid] = y0;
  out[base + tid + 256] = y1;
  if (outb) { outb[base + tid] = f2bf(y0); outb[base + tid + 256] = f2bf(y1); }
}

// --------------------------- final projection ------------------------------
__global__ void k_proj(const float* __restrict__ h, const float* __restrict__ pw,
                       const float* __restrict__ pb, float* __restrict__ out) {
  long long idx = (long long)blockIdx.x * blockDim.x + threadIdx.x;
  if (idx >= (long long)B_ * PRED_ * COUT_) return;
  const int c  = (int)(idx % COUT_);
  const int tp = (int)((idx / COUT_) % PRED_);
  const int b  = (int)(idx / ((long long)PRED_ * COUT_));
  const int t  = (N_ - PRED_) + tp;
  const float* hr = h + ((size_t)b * N_ + t) * D_;
  float acc = pb[c];
  for (int d = 0; d < D_; ++d) acc += hr[d] * pw[(size_t)d * COUT_ + c];
  out[idx] = acc;
}

// ---------------------------------------------------------------------------
extern "C" void kernel_launch(void* const* d_in, const int* in_sizes, int n_in,
                              void* d_out, int out_size, void* d_ws, size_t ws_size,
                              hipStream_t stream) {
  (void)in_sizes; (void)n_in; (void)out_size; (void)ws_size;
  const float* x_enc      = (const float*)d_in[0];
  const float* x_mark_enc = (const float*)d_in[1];
  const float* x_dec      = (const float*)d_in[2];
  const float* x_mark_dec = (const float*)d_in[3];
  const float* conv_w     = (const float*)d_in[4];
  const float* time_w     = (const float*)d_in[5];
  const float* wqk        = (const float*)d_in[6];
  const float* wv         = (const float*)d_in[7];
  const float* wo         = (const float*)d_in[8];
  const float* wob        = (const float*)d_in[9];
  const float* n1g        = (const float*)d_in[10];
  const float* n1b        = (const float*)d_in[11];
  const float* w1         = (const float*)d_in[12];
  const float* b1         = (const float*)d_in[13];
  const float* w2         = (const float*)d_in[14];
  const float* b2         = (const float*)d_in[15];
  const float* n2g        = (const float*)d_in[16];
  const float* n2b        = (const float*)d_in[17];
  const float* norm_g     = (const float*)d_in[18];
  const float* norm_b     = (const float*)d_in[19];
  const float* proj_w     = (const float*)d_in[20];
  const float* proj_b     = (const float*)d_in[21];
  const float* rotations  = (const float*)d_in[22];

  size_t off = 0;
  auto alloc = [&](size_t bytes) -> void* {
    void* p = (char*)d_ws + off;
    off += (bytes + 255) & ~(size_t)255;
    return p;
  };
  float*  f_h    = (float*)alloc((size_t)ROWS_ * D_ * 4);
  bf16_t* b_h    = (bf16_t*)alloc((size_t)ROWS_ * D_ * 2);
  float*  f_qk   = (float*)alloc((size_t)ROWS_ * D_ * 4);
  float*  f_v    = (float*)alloc((size_t)ROWS_ * D_ * 4);
  float*  f_attn = (float*)alloc((size_t)ROWS_ * D_ * 4);
  bf16_t* b_attn = (bf16_t*)alloc((size_t)ROWS_ * D_ * 2);
  float*  f_a    = (float*)alloc((size_t)ROWS_ * D_ * 4);
  float*  f_y    = (float*)alloc((size_t)ROWS_ * DFF_ * 4);
  bf16_t* b_y    = (bf16_t*)alloc((size_t)ROWS_ * DFF_ * 2);
  float*  f_y2   = (float*)alloc((size_t)ROWS_ * D_ * 4);
  int*    i_bkt  = (int*)alloc((size_t)BH_ * NJ_ * 4);
  int*    i_ord  = (int*)alloc((size_t)BH_ * NJ_ * 4);
  int*    i_und  = (int*)alloc((size_t)BH_ * NJ_ * 4);
  float*  f_slog = (float*)alloc((size_t)BH_ * NJ_ * 4);
  float*  f_so   = (float*)alloc((size_t)BH_ * NJ_ * DH_ * 4);
  bf16_t* b_wqk  = (bf16_t*)alloc((size_t)EL_ * D_ * D_ * 2);
  bf16_t* b_wv   = (bf16_t*)alloc((size_t)EL_ * D_ * D_ * 2);
  bf16_t* b_wo   = (bf16_t*)alloc((size_t)EL_ * D_ * D_ * 2);
  bf16_t* b_w1   = (bf16_t*)alloc((size_t)EL_ * D_ * DFF_ * 2);
  bf16_t* b_w2   = (bf16_t*)alloc((size_t)EL_ * DFF_ * D_ * 2);

  auto cvt = [&](const float* src, bf16_t* dst, long long n) {
    k_convert_bf16<<<(unsigned)((n + 255) / 256), 256, 0, stream>>>(src, dst, n);
  };
  cvt(wqk, b_wqk, (long long)EL_ * D_ * D_);
  cvt(wv,  b_wv,  (long long)EL_ * D_ * D_);
  cvt(wo,  b_wo,  (long long)EL_ * D_ * D_);
  cvt(w1,  b_w1,  (long long)EL_ * D_ * DFF_);
  cvt(w2,  b_w2,  (long long)EL_ * DFF_ * D_);

  {
    long long n = (long long)ROWS_ * D_;
    k_embed<<<(unsigned)((n + 255) / 256), 256, 0, stream>>>(
        x_enc, x_mark_enc, x_dec, x_mark_dec, conv_w, time_w, f_h, b_h);
  }

  const dim3 gD(D_ / BN_,   ROWS_ / BM_);    // (8, 360)
  const dim3 gF(DFF_ / BN_, ROWS_ / BM_);    // (32, 360)

  for (int l = 0; l < EL_; ++l) {
    const bf16_t* wqk_l = b_wqk + (size_t)l * D_ * D_;
    const bf16_t* wv_l  = b_wv  + (size_t)l * D_ * D_;
    const bf16_t* wo_l  = b_wo  + (size_t)l * D_ * D_;
    const bf16_t* w1_l  = b_w1  + (size_t)l * D_ * DFF_;
    const bf16_t* w2_l  = b_w2  + (size_t)l * DFF_ * D_;
    const float*  rot_l = rotations + (size_t)l * DH_ * NH_ * (NB_ / 2);

    k_gemm_bf16<<<gD, 256, 0, stream>>>(b_h, wqk_l, f_qk, nullptr, nullptr,
                                        ROWS_, D_, D_, 0);
    k_gemm_bf16<<<gD, 256, 0, stream>>>(b_h, wv_l, f_v, nullptr, nullptr,
                                        ROWS_, D_, D_, 0);

    {
      long long n = (long long)BH_ * NH_ * N_;
      k_buckets<<<(unsigned)((n + 255) / 256), 256, 0, stream>>>(f_qk, rot_l, i_bkt);
    }
    k_sort<<<BH_, 256, 0, stream>>>(i_bkt, i_ord, i_und);
    {
      long long n = (long long)BH_ * NJ_;
      k_attn<<<(unsigned)((n + 255) / 256), 256, 0, stream>>>(f_qk, f_v, i_ord,
                                                              f_so, f_slog);
    }
    {
      long long n = (long long)BH_ * N_;
      k_combine<<<(unsigned)((n + 255) / 256), 256, 0, stream>>>(f_so, f_slog, f_attn);
    }
    {
      long long n = (long long)ROWS_ * D_;
      k_convert_bf16<<<(unsigned)((n + 255) / 256), 256, 0, stream>>>(f_attn, b_attn, n);
    }

    k_gemm_bf16<<<gD, 256, 0, stream>>>(b_attn, wo_l, f_a, nullptr, wob + (size_t)l * D_,
                                        ROWS_, D_, D_, GF_BIAS);
    k_ln<<<ROWS_, 256, 0, stream>>>(f_h, f_a, n1g + (size_t)l * D_, n1b + (size_t)l * D_,
                                    f_h, b_h);

    k_gemm_bf16<<<gF, 256, 0, stream>>>(b_h, w1_l, f_y, b_y, b1 + (size_t)l * DFF_,
                                        ROWS_, DFF_, D_, GF_BIAS | GF_GELU | GF_BF16);
    k_gemm_bf16<<<gD, 256, 0, stream>>>(b_y, w2_l, f_y2, nullptr, b2 + (size_t)l * D_,
                                        ROWS_, D_, DFF_, GF_BIAS);
    k_ln<<<ROWS_, 256, 0, stream>>>(f_h, f_y2, n2g + (size_t)l * D_, n2b + (size_t)l * D_,
                                    f_h, b_h);
  }

  k_ln<<<ROWS_, 256, 0, stream>>>(f_h, nullptr, norm_g, norm_b, f_h, nullptr);
  {
    long long n = (long long)B_ * PRED_ * COUT_;
    k_proj<<<(unsigned)((n + 255) / 256), 256, 0, stream>>>(f_h, proj_w, proj_b,
                                                            (float*)d_out);
  }
}
